// Attention_83502754169400
// MI455X (gfx1250) — matrix operand order repaired and checked by reading, not compiled, not run on hardware
//
#include <hip/hip_runtime.h>
#include <hip/hip_bf16.h>
#include <math.h>

typedef __bf16 bf16_t;
typedef __attribute__((ext_vector_type(16))) __bf16 v16bf;
typedef __attribute__((ext_vector_type(8)))  __bf16 v8bf;
typedef __attribute__((ext_vector_type(4)))  __bf16 v4bf;
typedef __attribute__((ext_vector_type(8)))  float  v8f;
typedef __attribute__((ext_vector_type(4)))  float  v4f;
typedef __attribute__((ext_vector_type(4)))  int    v4i;

#define BATCH 2
#define SEQ   2048
#define DIM   2048
#define NH    16
#define HDIM  128
#define MROWS (BATCH*SEQ)

#define AS1 __attribute__((address_space(1)))
#define AS3 __attribute__((address_space(3)))

#if __has_builtin(__builtin_amdgcn_global_load_async_to_lds_b128)
#define USE_ASYNC_LDS 1
#else
#define USE_ASYNC_LDS 0
#endif

__device__ inline void wait_async0() {
#if __has_builtin(__builtin_amdgcn_s_wait_asynccnt)
  __builtin_amdgcn_s_wait_asynccnt(0);
#else
  asm volatile("s_wait_asynccnt 0x0" ::: "memory");
#endif
}

union FragBF { v16bf v; v8bf h[2]; };

// A-matrix 16x32 bf16 fragment (ISA 7.12.2): lane group g=lane/16, row=lane%16,
// elements 0..7 -> K = 8g..8g+7, elements 8..15 -> K = 16+8g..16+8g+7
__device__ inline v16bf frag_a(const bf16_t* row, int g) {
  FragBF f;
  f.h[0] = *(const v8bf*)(row + 8 * g);
  f.h[1] = *(const v8bf*)(row + 16 + 8 * g);
  return f.v;
}
// B-matrix 32x16 bf16 fragment: N=lane%16, elements 0..7 -> K = 8g..8g+7, 8..15 -> K = 16+8g..16+8g+7
__device__ inline v16bf frag_b(const bf16_t* row, int g) {
  FragBF f;
  f.h[0] = *(const v8bf*)(row + 8 * g);
  f.h[1] = *(const v8bf*)(row + 16 + 8 * g);
  return f.v;
}

__device__ inline v8f wmma_bf16(v16bf a, v16bf b, v8f c) {
  return __builtin_amdgcn_wmma_f32_16x16x32_bf16(false, a, false, b,
                                                 (short)0, c, false, false);
}

__device__ inline float red16_max(float v) {
#pragma unroll
  for (int m = 8; m >= 1; m >>= 1) v = fmaxf(v, __shfl_xor(v, m, 32));
  return v;
}
__device__ inline float red16_sum(float v) {
#pragma unroll
  for (int m = 8; m >= 1; m >>= 1) v += __shfl_xor(v, m, 32);
  return v;
}
__device__ inline float red32_sum(float v) {
#pragma unroll
  for (int m = 16; m >= 1; m >>= 1) v += __shfl_xor(v, m, 32);
  return v;
}

// ---------------------------------------------------------------------------
// fp32 -> bf16 elementwise convert (4-wide)
// ---------------------------------------------------------------------------
__global__ __launch_bounds__(256) void cvt_bf16_kernel(const float* __restrict__ x,
                                                       bf16_t* __restrict__ y) {
  const size_t i = ((size_t)blockIdx.x * blockDim.x + threadIdx.x) * 4;
  v4f v = *(const v4f*)(x + i);
  v4bf o;
#pragma unroll
  for (int j = 0; j < 4; ++j) o[j] = (bf16_t)v[j];
  *(v4bf*)(y + i) = o;
}

// ---------------------------------------------------------------------------
// W[k][n] fp32 -> Wt[n][k] bf16 (LDS-tiled transpose), DIM x DIM
// ---------------------------------------------------------------------------
__global__ __launch_bounds__(256) void transpose_w_kernel(const float* __restrict__ W,
                                                          bf16_t* __restrict__ Wt) {
  __shared__ float tile[32][33];
  const int tx = threadIdx.x & 31, ty = threadIdx.x >> 5;
  const int k0 = blockIdx.y * 32, n0 = blockIdx.x * 32;
#pragma unroll
  for (int i = 0; i < 4; ++i)
    tile[ty + 8 * i][tx] = W[(size_t)(k0 + ty + 8 * i) * DIM + n0 + tx];
  __syncthreads();
#pragma unroll
  for (int i = 0; i < 4; ++i)
    Wt[(size_t)(n0 + ty + 8 * i) * DIM + k0 + tx] = (bf16_t)tile[tx][ty + 8 * i];
}

// ---------------------------------------------------------------------------
// C[MROWS,DIM] = A[MROWS,DIM](bf16) x Wt[DIM,DIM](bf16, N-major)
// block tile 128x128, 8 waves (4 M x 2 N), each wave 32x64 = 2x4 WMMA tiles.
// Async path: GLOBAL_LOAD_ASYNC_TO_LDS_B128 double-buffered (ASYNCcnt),
// overlapping the HBM/L2 tile stream with the WMMA pipeline.
// ---------------------------------------------------------------------------
template <typename CT>
__global__ __launch_bounds__(256) void gemm_bf16_kernel(const bf16_t* __restrict__ A,
                                                        const bf16_t* __restrict__ Wt,
                                                        CT* __restrict__ C) {
  const int tid = threadIdx.x;
  const int lane = tid & 31, wave = tid >> 5;
  const int g = lane >> 4, ln = lane & 15;
  const int m0 = blockIdx.y * 128, n0 = blockIdx.x * 128;
  const int wm = (wave & 3) * 32, wn = (wave >> 2) * 64;
  const int lr = tid >> 1, lc = (tid & 1) * 16;

  v8f acc[2][4];
#pragma unroll
  for (int i = 0; i < 2; ++i)
#pragma unroll
    for (int j = 0; j < 4; ++j)
#pragma unroll
      for (int e = 0; e < 8; ++e) acc[i][j][e] = 0.f;

  const bf16_t* aptr = A + (size_t)(m0 + lr) * DIM + lc;
  const bf16_t* bptr = Wt + (size_t)(n0 + lr) * DIM + lc;

#if USE_ASYNC_LDS
  __shared__ __align__(16) bf16_t As[2][128 * 32];
  __shared__ __align__(16) bf16_t Bs[2][128 * 32];
  const int ldst = lr * 32 + lc;
  auto issue_tile = [&](int buf, int k0) {
    __builtin_amdgcn_global_load_async_to_lds_b128(
        (AS1 v4i*)(aptr + k0), (AS3 v4i*)&As[buf][ldst], 0, 0);
    __builtin_amdgcn_global_load_async_to_lds_b128(
        (AS1 v4i*)(aptr + k0 + 8), (AS3 v4i*)&As[buf][ldst + 8], 0, 0);
    __builtin_amdgcn_global_load_async_to_lds_b128(
        (AS1 v4i*)(bptr + k0), (AS3 v4i*)&Bs[buf][ldst], 0, 0);
    __builtin_amdgcn_global_load_async_to_lds_b128(
        (AS1 v4i*)(bptr + k0 + 8), (AS3 v4i*)&Bs[buf][ldst + 8], 0, 0);
  };
  issue_tile(0, 0);
  for (int k0 = 0; k0 < DIM; k0 += 32) {
    const int buf = (k0 >> 5) & 1;
    wait_async0();       // this wave's tile landed in LDS
    __syncthreads();     // every wave's tile landed
    if (k0 + 32 < DIM) issue_tile(buf ^ 1, k0 + 32);  // overlap next tile
    v16bf af[2], bfr[4];
#pragma unroll
    for (int mt = 0; mt < 2; ++mt) af[mt] = frag_a(&As[buf][(wm + mt * 16 + ln) * 32], g);
#pragma unroll
    for (int nt = 0; nt < 4; ++nt) bfr[nt] = frag_b(&Bs[buf][(wn + nt * 16 + ln) * 32], g);
#pragma unroll
    for (int mt = 0; mt < 2; ++mt)
#pragma unroll
      for (int nt = 0; nt < 4; ++nt) acc[mt][nt] = wmma_bf16(af[mt], bfr[nt], acc[mt][nt]);
  }
#else
  __shared__ __align__(16) bf16_t As[128 * 32];
  __shared__ __align__(16) bf16_t Bs[128 * 32];
  for (int k0 = 0; k0 < DIM; k0 += 32) {
    __syncthreads();
    *(v8bf*)&As[lr * 32 + lc]     = *(const v8bf*)(aptr + k0);
    *(v8bf*)&As[lr * 32 + lc + 8] = *(const v8bf*)(aptr + k0 + 8);
    *(v8bf*)&Bs[lr * 32 + lc]     = *(const v8bf*)(bptr + k0);
    *(v8bf*)&Bs[lr * 32 + lc + 8] = *(const v8bf*)(bptr + k0 + 8);
    __syncthreads();
    if (k0 + 32 < DIM) {
      __builtin_prefetch(aptr + k0 + 32, 0, 1);  // global_prefetch_b8
      __builtin_prefetch(bptr + k0 + 32, 0, 1);
    }
    v16bf af[2], bfr[4];
#pragma unroll
    for (int mt = 0; mt < 2; ++mt) af[mt] = frag_a(&As[(wm + mt * 16 + ln) * 32], g);
#pragma unroll
    for (int nt = 0; nt < 4; ++nt) bfr[nt] = frag_b(&Bs[(wn + nt * 16 + ln) * 32], g);
#pragma unroll
    for (int mt = 0; mt < 2; ++mt)
#pragma unroll
      for (int nt = 0; nt < 4; ++nt) acc[mt][nt] = wmma_bf16(af[mt], bfr[nt], acc[mt][nt]);
  }
#endif

#pragma unroll
  for (int mt = 0; mt < 2; ++mt)
#pragma unroll
    for (int nt = 0; nt < 4; ++nt)
#pragma unroll
      for (int e = 0; e < 8; ++e) {
        const int row = m0 + wm + mt * 16 + e + 8 * g;
        const int col = n0 + wn + nt * 16 + ln;
        C[(size_t)row * DIM + col] = (CT)acc[mt][nt][e];
      }
}

// ---------------------------------------------------------------------------
// Fused per-head LayerNorm + RoPE + repack.
// One wave per (b,s,h): 32 lanes x 4 elements = HD 128.
// Qp/Kp: [B,H,S,HD] bf16.  Vp: [B,H,HD,S] bf16 (transposed for attention B-frags)
// ---------------------------------------------------------------------------
__global__ __launch_bounds__(256) void ln_rope_pack_kernel(
    const bf16_t* __restrict__ Qr, const bf16_t* __restrict__ Kr,
    const bf16_t* __restrict__ Vr, const float* __restrict__ fcos,
    const float* __restrict__ fsin, const float* __restrict__ qw,
    const float* __restrict__ qbias, const float* __restrict__ kw,
    const float* __restrict__ kbias, bf16_t* __restrict__ Qp,
    bf16_t* __restrict__ Kp, bf16_t* __restrict__ Vp) {
  const int gtid = blockIdx.x * blockDim.x + threadIdx.x;
  const int lane = gtid & 31;
  const int wid = gtid >> 5;  // 0 .. B*S*H-1
  const int h = wid % NH;
  const int s = (wid / NH) % SEQ;
  const int b = wid / (NH * SEQ);
  const int hd0 = lane * 4;

  const size_t rawoff = ((size_t)(b * SEQ + s)) * DIM + h * HDIM + hd0;
  v4bf q4 = *(const v4bf*)(Qr + rawoff);
  v4bf k4 = *(const v4bf*)(Kr + rawoff);
  v4bf v4 = *(const v4bf*)(Vr + rawoff);
  float q[4], k[4];
#pragma unroll
  for (int i = 0; i < 4; ++i) { q[i] = (float)q4[i]; k[i] = (float)k4[i]; }

  // LayerNorm over HD=128 (wave-wide reductions)
  {
    float mu = red32_sum(q[0] + q[1] + q[2] + q[3]) * (1.0f / HDIM);
    float vs = 0.f;
#pragma unroll
    for (int i = 0; i < 4; ++i) { float d = q[i] - mu; vs += d * d; }
    float r = rsqrtf(red32_sum(vs) * (1.0f / HDIM) + 1e-5f);
#pragma unroll
    for (int i = 0; i < 4; ++i) q[i] = (q[i] - mu) * r * qw[hd0 + i] + qbias[hd0 + i];
  }
  {
    float mu = red32_sum(k[0] + k[1] + k[2] + k[3]) * (1.0f / HDIM);
    float vs = 0.f;
#pragma unroll
    for (int i = 0; i < 4; ++i) { float d = k[i] - mu; vs += d * d; }
    float r = rsqrtf(red32_sum(vs) * (1.0f / HDIM) + 1e-5f);
#pragma unroll
    for (int i = 0; i < 4; ++i) k[i] = (k[i] - mu) * r * kw[hd0 + i] + kbias[hd0 + i];
  }

  // RoPE: lane covers pairs p0=2*lane, p1=2*lane+1
  const int fidx = s * (HDIM / 2) + lane * 2;
  const float c0 = fcos[fidx], c1 = fcos[fidx + 1];
  const float sn0 = fsin[fidx], sn1 = fsin[fidx + 1];
  v4bf qo, ko;
  qo[0] = (bf16_t)(q[0] * c0 - q[1] * sn0);
  qo[1] = (bf16_t)(q[0] * sn0 + q[1] * c0);
  qo[2] = (bf16_t)(q[2] * c1 - q[3] * sn1);
  qo[3] = (bf16_t)(q[2] * sn1 + q[3] * c1);
  ko[0] = (bf16_t)(k[0] * c0 - k[1] * sn0);
  ko[1] = (bf16_t)(k[0] * sn0 + k[1] * c0);
  ko[2] = (bf16_t)(k[2] * c1 - k[3] * sn1);
  ko[3] = (bf16_t)(k[2] * sn1 + k[3] * c1);

  const size_t pkoff = (((size_t)(b * NH + h)) * SEQ + s) * HDIM + hd0;
  *(v4bf*)(Qp + pkoff) = qo;
  *(v4bf*)(Kp + pkoff) = ko;

  const size_t vbase = ((size_t)(b * NH + h)) * HDIM;
#pragma unroll
  for (int i = 0; i < 4; ++i) Vp[(vbase + hd0 + i) * SEQ + s] = v4[i];
}

// ---------------------------------------------------------------------------
// Causal flash attention.  Block = 4 waves x 16 q-rows = 64 q rows per (b,h).
// Q/K: [B,H,S,HD] bf16, Vt: [B,H,HD,S] bf16, O: [B,S,H,HD] bf16.
// K/V stream from global (bf16 K+V per head set fits in 192MB L2).
// ---------------------------------------------------------------------------
__global__ __launch_bounds__(128) void flash_attn_kernel(const bf16_t* __restrict__ Q,
                                                         const bf16_t* __restrict__ K,
                                                         const bf16_t* __restrict__ Vt,
                                                         bf16_t* __restrict__ O) {
  __shared__ __align__(16) bf16_t Pld[4][16][32];
  const int lane = threadIdx.x & 31, wave = threadIdx.x >> 5;
  const int g = lane >> 4, ln = lane & 15;
  const int q0 = blockIdx.x * 64;
  const int bh = blockIdx.y;
  const int b = bh / NH, h = bh % NH;
  const bf16_t* Qh = Q + (size_t)bh * SEQ * HDIM;
  const bf16_t* Kh = K + (size_t)bh * SEQ * HDIM;
  const bf16_t* Vh = Vt + (size_t)bh * HDIM * SEQ;
  const float scale = 0.08838834764831845f;  // 1/sqrt(HD)

  const int qa = q0 + wave * 16 + ln;  // this lane's A-fragment row
  v16bf qf[4];
#pragma unroll
  for (int j = 0; j < 4; ++j) qf[j] = frag_a(Qh + (size_t)qa * HDIM + j * 32, g);

  v8f o[8];
#pragma unroll
  for (int t = 0; t < 8; ++t)
#pragma unroll
    for (int e = 0; e < 8; ++e) o[t][e] = 0.f;
  float mi[8], li[8];
#pragma unroll
  for (int e = 0; e < 8; ++e) { mi[e] = -1e30f; li[e] = 0.f; }

  const int rbase = q0 + wave * 16 + 8 * g;  // C-fragment row base for this lane

  for (int k0 = 0; k0 < q0 + 64; k0 += 32) {
    v8f s0, s1;
#pragma unroll
    for (int e = 0; e < 8; ++e) { s0[e] = 0.f; s1[e] = 0.f; }
    // scores: Q(16x128) x K^T(128x32) via 4 WMMAs per 16-key half
#pragma unroll
    for (int j = 0; j < 4; ++j) {
      v16bf kf0 = frag_b(Kh + (size_t)(k0 + ln) * HDIM + j * 32, g);
      s0 = wmma_bf16(qf[j], kf0, s0);
      v16bf kf1 = frag_b(Kh + (size_t)(k0 + 16 + ln) * HDIM + j * 32, g);
      s1 = wmma_bf16(qf[j], kf1, s1);
    }
    // online softmax (rows = e + 8g, cols = ln / 16+ln)
    float alpha[8];
#pragma unroll
    for (int e = 0; e < 8; ++e) {
      const int rg = rbase + e;
      float a0 = s0[e] * scale; if (k0 + ln > rg) a0 = -1e9f;
      float a1 = s1[e] * scale; if (k0 + 16 + ln > rg) a1 = -1e9f;
      const float mx = red16_max(fmaxf(a0, a1));
      const float mn = fmaxf(mi[e], mx);
      alpha[e] = __expf(mi[e] - mn);
      const float p0 = __expf(a0 - mn);
      const float p1 = __expf(a1 - mn);
      s0[e] = p0; s1[e] = p1;
      li[e] = li[e] * alpha[e] + red16_sum(p0 + p1);
      mi[e] = mn;
    }
#pragma unroll
    for (int t = 0; t < 8; ++t)
#pragma unroll
      for (int e = 0; e < 8; ++e) o[t][e] *= alpha[e];
    // re-layout P: C-fragment -> A-fragment via per-wave LDS (same-wave DS order)
#pragma unroll
    for (int e = 0; e < 8; ++e) {
      Pld[wave][e + 8 * g][ln]      = (bf16_t)s0[e];
      Pld[wave][e + 8 * g][16 + ln] = (bf16_t)s1[e];
    }
    v16bf pf = frag_a(&Pld[wave][ln][0], g);
    // O += P(16x32) x V(32x128): V^T fragments contiguous along keys
#pragma unroll
    for (int t = 0; t < 8; ++t) {
      v16bf vf = frag_b(Vh + (size_t)(t * 16 + ln) * SEQ + k0, g);
      o[t] = wmma_bf16(pf, vf, o[t]);
    }
  }
  // epilogue: normalize and store [B,S,H,HD]
#pragma unroll
  for (int t = 0; t < 8; ++t)
#pragma unroll
    for (int e = 0; e < 8; ++e) {
      const int rg = rbase + e;
      const int hd = t * 16 + ln;
      const float val = o[t][e] / li[e];
      O[(((size_t)b * SEQ + rg) * NH + h) * HDIM + hd] = (bf16_t)val;
    }
}

// ---------------------------------------------------------------------------
extern "C" void kernel_launch(void* const* d_in, const int* in_sizes, int n_in,
                              void* d_out, int out_size, void* d_ws, size_t ws_size,
                              hipStream_t stream) {
  const float* x    = (const float*)d_in[0];
  const float* fcos = (const float*)d_in[1];
  const float* fsin = (const float*)d_in[2];
  // d_in[3] = mask: causal, handled analytically
  const float* wq  = (const float*)d_in[4];
  const float* wk  = (const float*)d_in[5];
  const float* wv  = (const float*)d_in[6];
  const float* wo  = (const float*)d_in[7];
  const float* qlw = (const float*)d_in[8];
  const float* qlb = (const float*)d_in[9];
  const float* klw = (const float*)d_in[10];
  const float* klb = (const float*)d_in[11];
  float* out = (float*)d_out;

  char* ws = (char*)d_ws;
  size_t off = 0;
  auto alloc = [&](size_t bytes) -> void* {
    void* p = ws + off;
    off += (bytes + 255) & ~(size_t)255;
    return p;
  };
  const size_t actsz = (size_t)MROWS * DIM * sizeof(bf16_t);
  const size_t wsz   = (size_t)DIM * DIM * sizeof(bf16_t);
  bf16_t* xb  = (bf16_t*)alloc(actsz);
  bf16_t* wqt = (bf16_t*)alloc(wsz);
  bf16_t* wkt = (bf16_t*)alloc(wsz);
  bf16_t* wvt = (bf16_t*)alloc(wsz);
  bf16_t* wot = (bf16_t*)alloc(wsz);
  bf16_t* Qr  = (bf16_t*)alloc(actsz);
  bf16_t* Kr  = (bf16_t*)alloc(actsz);
  bf16_t* Vr  = (bf16_t*)alloc(actsz);
  bf16_t* Qp  = (bf16_t*)alloc(actsz);
  bf16_t* Kp  = (bf16_t*)alloc(actsz);
  bf16_t* Vp  = (bf16_t*)alloc(actsz);
  bf16_t* Ob  = xb;  // xb dead after QKV GEMMs; reuse for attention output

  // 1. x -> bf16
  cvt_bf16_kernel<<<(MROWS * DIM / 4) / 256, 256, 0, stream>>>(x, xb);
  // 2. weights -> bf16, transposed to [N][K]
  dim3 tg(DIM / 32, DIM / 32);
  transpose_w_kernel<<<tg, 256, 0, stream>>>(wq, wqt);
  transpose_w_kernel<<<tg, 256, 0, stream>>>(wk, wkt);
  transpose_w_kernel<<<tg, 256, 0, stream>>>(wv, wvt);
  transpose_w_kernel<<<tg, 256, 0, stream>>>(wo, wot);
  // 3. QKV projections (WMMA, async double-buffered LDS tiles)
  dim3 gg(DIM / 128, MROWS / 128);
  gemm_bf16_kernel<bf16_t><<<gg, 256, 0, stream>>>(xb, wqt, Qr);
  gemm_bf16_kernel<bf16_t><<<gg, 256, 0, stream>>>(xb, wkt, Kr);
  gemm_bf16_kernel<bf16_t><<<gg, 256, 0, stream>>>(xb, wvt, Vr);
  // 4. LayerNorm + RoPE + repack (one wave per (b,s,h))
  ln_rope_pack_kernel<<<(MROWS * NH * 32) / 256, 256, 0, stream>>>(
      Qr, Kr, Vr, fcos, fsin, qlw, qlb, klw, klb, Qp, Kp, Vp);
  // 5. causal flash attention (WMMA)
  dim3 fg(SEQ / 64, BATCH * NH);
  flash_attn_kernel<<<fg, 128, 0, stream>>>(Qp, Kp, Vp, Ob);
  // 6. output projection (WMMA, fp32 out)
  gemm_bf16_kernel<float><<<gg, 256, 0, stream>>>(Ob, wot, out);
  (void)in_sizes; (void)n_in; (void)out_size; (void)ws_size;
}
